// Inference_85315230368232
// MI455X (gfx1250) — matrix operand order repaired and checked by reading, not compiled, not run on hardware
//
#include <hip/hip_runtime.h>
#include <cstdint>
#include <cstddef>

// ---------------------------------------------------------------------------
// Shifted-window displacement attention for MI455X (gfx1250, wave32, WMMA).
// One workgroup (256 threads = 8 waves) per (window, head):
//   grid = (1024 windows, 4 heads).  All five contractions run on
//   v_wmma_f32_16x16x32_f16; rpe einsums are converted to GEMM+gather via the
//   relative-index identity r(i,j) = (dh+7)*15 + (dw+7).
// ---------------------------------------------------------------------------

#define WS 8
#define SHIFT 4
#define HEADS 4
#define HEAD_DIM 32
#define N_DISP 2
#define M_TOK 128            // tokens per window
#define NUM_WIN 256          // windows per image
#define NRP 225              // (2*8-1)^2 relative positions
#define NRP_PAD 240          // padded to 15 N-tiles of 16
#define K_PAD 256            // K padding for S @ Tv
#define SCALE_F 0.1767766952966369f   // 32^-0.5

typedef __attribute__((ext_vector_type(16))) _Float16 v16h;
typedef __attribute__((ext_vector_type(8)))  _Float16 v8h;
typedef __attribute__((ext_vector_type(8)))  float    v8f;

// LDS layout (element offsets, _Float16 units)
#define OFF_QS   0                      // q scaled,   [128][32]
#define OFF_KS   (OFF_QS  + 128*32)     // k,          [128][32]
#define OFF_VTS  (OFF_KS  + 128*32)     // v^T,        [32][128]
#define OFF_TK   (OFF_VTS + 32*128)     // k_embed tab [240][32]
#define OFF_TQ   (OFF_TK  + 240*32)     // q_embed tab [240][32] (scaled)
#define OFF_TVT  (OFF_TQ  + 240*32)     // v_embed^T   [32][256]
#define OFF_QR   (OFF_TVT + 32*256)     // Qr col-maj  [240][128]
#define OFF_KR   (OFF_QR  + 240*128)    // Kr col-maj  [240][128]
#define OFF_AT   (OFF_KR  + 240*128)    // attn probs  [128][128]
#define OFF_S    (OFF_AT  + 128*128)    // scattered S [128][256]
#define LDS_HALVES (OFF_S + 128*256)    // 146432 halves = 292864 bytes
#define LDS_BYTES  (LDS_HALVES * 2)

// A fragment (16x32 f16, row-major source, stride in elements).
// lane<16 : row = lane,      K = k0+0..7  and k0+16..23
// lane>=16: row = lane-16,   K = k0+8..15 and k0+24..31
__device__ __forceinline__ v16h load_a_frag(const _Float16* base, int stride,
                                            int row0, int k0, int lane) {
  const int r  = row0 + (lane & 15);
  const int hs = lane >> 4;
  const _Float16* p = base + r * stride + k0 + hs * 8;
  v8h lo = *(const v8h*)(p);
  v8h hi = *(const v8h*)(p + 16);
  return __builtin_shufflevector(lo, hi, 0,1,2,3,4,5,6,7,8,9,10,11,12,13,14,15);
}

// B fragment (32x16 f16). Source is B^T stored row-major: Bt[n][k].
// lane<16 : col = lane,     K = k0+0..7 and k0+16..23 ; lane>=16: col = lane-16, K = k0+8..15 and k0+24..31
__device__ __forceinline__ v16h load_b_frag(const _Float16* base, int stride,
                                            int n0, int k0, int lane) {
  const int n = n0 + (lane & 15);
  const int k = k0 + (lane >> 4) * 8;
  const _Float16* p = base + n * stride + k;
  v8h lo = *(const v8h*)(p);
  v8h hi = *(const v8h*)(p + 16);
  return __builtin_shufflevector(lo, hi, 0,1,2,3,4,5,6,7,8,9,10,11,12,13,14,15);
}

__device__ __forceinline__ v8f wmma16(v16h a, v16h b, v8f c) {
  return __builtin_amdgcn_wmma_f32_16x16x32_f16(false, a, false, b,
                                                (short)0, c, false, false);
}

__global__ __launch_bounds__(256, 1)
void disp_swin_attn(const float* __restrict__ qkv,
                    const float* __restrict__ mask,
                    const float* __restrict__ rpe,
                    float* __restrict__ out) {
  extern __shared__ _Float16 sm[];
  _Float16* qs   = sm + OFF_QS;
  _Float16* ks   = sm + OFF_KS;
  _Float16* vts  = sm + OFF_VTS;
  _Float16* Tk   = sm + OFF_TK;
  _Float16* Tq   = sm + OFF_TQ;
  _Float16* Tvt  = sm + OFF_TVT;
  _Float16* Qr   = sm + OFF_QR;
  _Float16* Kr   = sm + OFF_KR;
  _Float16* attS = sm + OFF_AT;
  _Float16* Smat = sm + OFF_S;

  const int tid  = threadIdx.x;
  const int lane = tid & 31;
  const int wv   = tid >> 5;        // wave id = attention row-tile (0..7)
  const int hs   = lane >> 4;       // lane half (selects M row group)
  const int lcol = lane & 15;       // N index within tile

  const int g   = blockIdx.x;       // global window 0..1023
  const int hd  = blockIdx.y;       // head 0..3
  const int b   = g >> 8;
  const int win = g & 255;
  const int wi  = win >> 4;
  const int wj  = win & 15;

  // ------------------------------------------------------------------ load
  {
    // q/k/v for this (window, head): token = tid>>1, channel half = tid&1
    const int t  = tid >> 1;
    const int c0 = (tid & 1) * 16;
    const int n  = t & 1, pp = t >> 1, wh = pp >> 3, ww = pp & 7;
    const int gr = (wi * WS + wh + SHIFT) & 127;   // roll(-SHIFT) folded in
    const int gc = (wj * WS + ww + SHIFT) & 127;
    const size_t base = ((((size_t)b * 128 + gr) * 128 + gc) * N_DISP + n) * 384;
    const float4* qp = (const float4*)(qkv + base +       hd * 32 + c0);
    const float4* kp = (const float4*)(qkv + base + 128 + hd * 32 + c0);
    const float4* vp = (const float4*)(qkv + base + 256 + hd * 32 + c0);
#pragma unroll
    for (int u = 0; u < 4; ++u) {
      float4 qv = qp[u], kv = kp[u], vv = vp[u];
      const int c = c0 + u * 4;
      qs[t * 32 + c + 0] = (_Float16)(qv.x * SCALE_F);
      qs[t * 32 + c + 1] = (_Float16)(qv.y * SCALE_F);
      qs[t * 32 + c + 2] = (_Float16)(qv.z * SCALE_F);
      qs[t * 32 + c + 3] = (_Float16)(qv.w * SCALE_F);
      ks[t * 32 + c + 0] = (_Float16)kv.x;
      ks[t * 32 + c + 1] = (_Float16)kv.y;
      ks[t * 32 + c + 2] = (_Float16)kv.z;
      ks[t * 32 + c + 3] = (_Float16)kv.w;
      vts[(c + 0) * 128 + t] = (_Float16)vv.x;
      vts[(c + 1) * 128 + t] = (_Float16)vv.y;
      vts[(c + 2) * 128 + t] = (_Float16)vv.z;
      vts[(c + 3) * 128 + t] = (_Float16)vv.w;
    }
  }
  // warm the per-window mask tile into cache (global_prefetch_b8)
  __builtin_prefetch(mask + (size_t)win * (M_TOK * M_TOK) + tid * 64, 0, 0);

  {
    // relative-position tables for this head: q/k/v embed slices, padded
    const int r = tid;                        // 0..255
    if (r < NRP_PAD) {
      if (r < NRP) {
        const float* tp = rpe + (size_t)r * 384 + hd * 96;
#pragma unroll 8
        for (int c = 0; c < 32; ++c) {
          Tq[r * 32 + c] = (_Float16)(tp[c] * SCALE_F);   // q_embed (scaled)
          Tk[r * 32 + c] = (_Float16)tp[32 + c];          // k_embed
        }
      } else {
#pragma unroll 8
        for (int c = 0; c < 32; ++c) { Tq[r * 32 + c] = (_Float16)0.f;
                                       Tk[r * 32 + c] = (_Float16)0.f; }
      }
    }
    if (r < NRP) {
      const float* tp = rpe + (size_t)r * 384 + hd * 96 + 64;
#pragma unroll 8
      for (int c = 0; c < 32; ++c) Tvt[c * K_PAD + r] = (_Float16)tp[c];
    } else {
#pragma unroll 8
      for (int c = 0; c < 32; ++c) Tvt[c * K_PAD + r] = (_Float16)0.f;
    }
  }
  { // zero the scatter matrix S [128][256] f16
    uint4 z = {0u, 0u, 0u, 0u};
    uint4* sp = (uint4*)Smat;
    for (int i = tid; i < (128 * K_PAD * 2) / 16; i += 256) sp[i] = z;
  }
  __syncthreads();

  // -------------------------------------------------- Qr = q*Tk^T, Kr = k*Tq^T
  const v16h aQ = load_a_frag(qs, 32, wv * 16, 0, lane);
  const v16h aK = load_a_frag(ks, 32, wv * 16, 0, lane);
  const int rowb = wv * 16 + hs * 8;
  for (int ct = 0; ct < 15; ++ct) {
    v8f dq = {}; v8f dk = {};
    dq = wmma16(aQ, load_b_frag(Tk, 32, ct * 16, 0, lane), dq);
    dk = wmma16(aK, load_b_frag(Tq, 32, ct * 16, 0, lane), dk);
    const int colb = ct * 16 + lcol;
    v8h hq, hk;
#pragma unroll
    for (int v = 0; v < 8; ++v) { hq[v] = (_Float16)dq[v]; hk[v] = (_Float16)dk[v]; }
    *(v8h*)(Qr + colb * 128 + rowb) = hq;   // column-major: 1 ds_store_b128
    *(v8h*)(Kr + colb * 128 + rowb) = hk;
  }
  __syncthreads();

  // -------------------------------------------- logits = qk + qr + kr + mask
  v8f acc[8];
#pragma unroll
  for (int ct = 0; ct < 8; ++ct) {
    v8f c0 = {};
    acc[ct] = wmma16(aQ, load_b_frag(ks, 32, ct * 16, 0, lane), c0);
  }
  const float* mp = mask + (size_t)win * (M_TOK * M_TOK);
#pragma unroll
  for (int v = 0; v < 8; ++v) {
    const int row = wv * 16 + hs * 8 + v;
    const int i64 = row >> 1;
    const int ih = i64 >> 3, iw = i64 & 7;
    float mx = -3.402823466e38f;
    float lg[8];
#pragma unroll
    for (int ct = 0; ct < 8; ++ct) {
      const int col = ct * 16 + lcol;
      const int j64 = col >> 1;
      const int ridx = (ih - (j64 >> 3) + 7) * 15 + (iw - (j64 & 7) + 7);
      float l = acc[ct][v] + (float)Qr[ridx * 128 + row]
                           + (float)Kr[ridx * 128 + col]
                           + mp[row * 128 + col];
      lg[ct] = l;
      mx = fmaxf(mx, l);
    }
#pragma unroll
    for (int off = 1; off < 16; off <<= 1) mx = fmaxf(mx, __shfl_xor(mx, off));
    float sum = 0.f;
#pragma unroll
    for (int ct = 0; ct < 8; ++ct) { float e = __expf(lg[ct] - mx); lg[ct] = e; sum += e; }
#pragma unroll
    for (int off = 1; off < 16; off <<= 1) sum += __shfl_xor(sum, off);
    const float inv = 1.0f / sum;
#pragma unroll
    for (int ct = 0; ct < 8; ++ct) {
      const float pr = lg[ct] * inv;
      const int col = ct * 16 + lcol;
      attS[row * 128 + col] = (_Float16)pr;
      const float pair = pr + __shfl_down(pr, 1);   // sum the 2 candidates
      if ((lcol & 1) == 0) {                        // j64 -> r is injective per row
        const int j64 = col >> 1;
        const int ridx = (ih - (j64 >> 3) + 7) * 15 + (iw - (j64 & 7) + 7);
        Smat[row * K_PAD + ridx] = (_Float16)pair;
      }
    }
  }
  __syncthreads();

  // ------------------------------------------- out = attn*v + S*Tv (GEMMs)
  v8f o0 = {}, o1 = {};
#pragma unroll
  for (int kk = 0; kk < 4; ++kk) {
    v16h a = load_a_frag(attS, 128, wv * 16, kk * 32, lane);
    o0 = wmma16(a, load_b_frag(vts, 128, 0,  kk * 32, lane), o0);
    o1 = wmma16(a, load_b_frag(vts, 128, 16, kk * 32, lane), o1);
  }
#pragma unroll
  for (int kk = 0; kk < 8; ++kk) {
    v16h a = load_a_frag(Smat, K_PAD, wv * 16, kk * 32, lane);
    o0 = wmma16(a, load_b_frag(Tvt, K_PAD, 0,  kk * 32, lane), o0);
    o1 = wmma16(a, load_b_frag(Tvt, K_PAD, 16, kk * 32, lane), o1);
  }
  // un-partition + roll(+SHIFT) folded into the store index
#pragma unroll
  for (int v = 0; v < 8; ++v) {
    const int t  = wv * 16 + hs * 8 + v;
    const int n  = t & 1, pp = t >> 1, wh = pp >> 3, ww = pp & 7;
    const int gr = (wi * WS + wh + SHIFT) & 127;
    const int gc = (wj * WS + ww + SHIFT) & 127;
    float* op = out + ((((size_t)b * 128 + gr) * 128 + gc) * N_DISP + n) * 128
                    + hd * 32;
    op[lcol]      = o0[v];
    op[16 + lcol] = o1[v];
  }
}

extern "C" void kernel_launch(void* const* d_in, const int* in_sizes, int n_in,
                              void* d_out, int out_size, void* d_ws, size_t ws_size,
                              hipStream_t stream) {
  (void)in_sizes; (void)n_in; (void)out_size; (void)d_ws; (void)ws_size;
  const float* qkv  = (const float*)d_in[0];
  const float* mask = (const float*)d_in[1];
  const float* rpe  = (const float*)d_in[2];
  // d_in[3] (rel_pos_index) unused: r(i,j) = (dh+7)*15 + (dw+7) computed inline
  float* out = (float*)d_out;

  (void)hipFuncSetAttribute((const void*)disp_swin_attn,
                            hipFuncAttributeMaxDynamicSharedMemorySize,
                            LDS_BYTES);
  dim3 grid(1024, HEADS);
  dim3 block(256);
  disp_swin_attn<<<grid, block, LDS_BYTES, stream>>>(qkv, mask, rpe, out);
}
